// RMNE_83502754169132
// MI455X (gfx1250) — compile-verified, matrix operand order checked by reading, not run on hardware
//
#include <hip/hip_runtime.h>

// Problem constants (match reference)
#define NNET   2
#define VSZ    100000
#define DIM    128
#define BATCH  4096
#define KNB    5
#define NS     10
#define NTERMS 20

// Launch shape
#define NBLK 128   // x-blocks per term
#define TPB  256   // 8 waves of 32

typedef float    v2f  __attribute__((ext_vector_type(2)));
typedef float    v8f  __attribute__((ext_vector_type(8)));
typedef _Float16 v16h __attribute__((ext_vector_type(16)));

struct Job {
  const float* ctx_table;   // table the ctx rows are gathered from
  const int*   idx;         // B x M gather indices
  const float* node_table;  // node_tables[i]
  const int*   node_idx;    // nodes_idx[i] (B)
  int   M;                  // samples per batch element
  int   negate;             // 1 -> logsigmoid(-s)
  float coef;               // -w / (10*B*(pos?M:1))
  int   pad;
};
struct JobArgs { Job jobs[NTERMS]; };

// Layout-agnostic wave32 sum via WMMA with an all-ones B matrix:
// D[m,n] = sum_k A[m,k]; summing one lane's 8 accumulator VGPRs gives the
// row-sums of one half of the rows; adding the other lane-half (shfl_xor 16)
// gives sum over ALL 64 A-register slots = sum over all lanes.
__device__ __forceinline__ float wave_sum_wmma(float v) {
#if __has_builtin(__builtin_amdgcn_wmma_f32_16x16x4_f32)
  v2f a;    a[0] = v;     a[1] = 0.0f;
  v2f ones; ones[0] = 1.0f; ones[1] = 1.0f;
  v8f c = {};
  c = __builtin_amdgcn_wmma_f32_16x16x4_f32(false, a, false, ones,
                                            (short)0, c, false, false);
#else
  v16h a = {};
  a[0] = (_Float16)v;
  v16h ones;
#pragma unroll
  for (int i = 0; i < 16; ++i) ones[i] = (_Float16)1.0f;
  v8f c = {};
  c = __builtin_amdgcn_wmma_f32_16x16x32_f16(false, a, false, ones,
                                             (short)0, c, false, false);
#endif
  float t = 0.0f;
#pragma unroll
  for (int i = 0; i < 8; ++i) t += c[i];
  t += __shfl_xor(t, 16, 32);   // combine the two lane-half row sets
  return t;                     // every lane holds the wave sum
}

__device__ __forceinline__ float log_sigmoid(float x) {
  // stable: min(x,0) - log(1 + exp(-|x|))
  return fminf(x, 0.0f) - __logf(1.0f + __expf(-fabsf(x)));
}

__global__ void __launch_bounds__(TPB)
terms_kernel(JobArgs args, float* __restrict__ partials) {
  const Job jb    = args.jobs[blockIdx.y];
  const int M     = jb.M;
  const int total = BATCH * M;

  float acc = 0.0f;
  for (int e = blockIdx.x * TPB + threadIdx.x; e < total; e += NBLK * TPB) {
    const int b = e / M;
    const int m = e - b * M;

    const float4* c4 =
        (const float4*)(jb.ctx_table + (size_t)jb.idx[(size_t)b * M + m] * DIM);
    const float4* n4 =
        (const float4*)(jb.node_table + (size_t)jb.node_idx[b] * DIM);

    // Prefetch next gathered ctx row (global_prefetch_b8).
    const int e2 = e + NBLK * TPB;
    if (e2 < total) {
      const int b2 = e2 / M;
      const int m2 = e2 - b2 * M;
      __builtin_prefetch(
          jb.ctx_table + (size_t)jb.idx[(size_t)b2 * M + m2] * DIM, 0, 1);
    }

    // 512B contiguous row per lane: 32x global_load_b128 pairs, fully
    // consumed cachelines, L2-resident after first touch (tables ~= 192MB L2)
    float s = 0.0f;
#pragma unroll 8
    for (int d = 0; d < DIM / 4; ++d) {
      const float4 cv = c4[d];
      const float4 nv = n4[d];
      s = fmaf(cv.x, nv.x, s);
      s = fmaf(cv.y, nv.y, s);
      s = fmaf(cv.z, nv.z, s);
      s = fmaf(cv.w, nv.w, s);
    }

    const float x = jb.negate ? -s : s;
    acc += jb.coef * log_sigmoid(x);
  }

  // Uniform control flow here: EXEC all-ones as WMMA requires.
  const float wsum = wave_sum_wmma(acc);

  __shared__ float red[TPB / 32];
  const int lane = threadIdx.x & 31;
  const int wid  = threadIdx.x >> 5;
  if (lane == 0) red[wid] = wsum;
  __syncthreads();
  if (threadIdx.x == 0) {
    float t = 0.0f;
#pragma unroll
    for (int i = 0; i < TPB / 32; ++i) t += red[i];
    partials[(size_t)blockIdx.y * NBLK + blockIdx.x] = t;
  }
}

__global__ void __launch_bounds__(32)
finalize_kernel(const float* __restrict__ partials, float* __restrict__ out) {
  float acc = 0.0f;
  for (int i = threadIdx.x; i < NTERMS * NBLK; i += 32) acc += partials[i];
  const float t = wave_sum_wmma(acc);
  if (threadIdx.x == 0) out[0] = t;
}

extern "C" void kernel_launch(void* const* d_in, const int* in_sizes, int n_in,
                              void* d_out, int out_size, void* d_ws,
                              size_t ws_size, hipStream_t stream) {
  (void)in_sizes; (void)n_in; (void)out_size; (void)ws_size;

  const float* node_tables  = (const float*)d_in[0];  // (2,V,D)
  const float* neigh_tables = (const float*)d_in[1];  // (2,V,D)
  const int*   nodes_idx    = (const int*)d_in[2];    // (2,B)
  const int*   neigh_idx    = (const int*)d_in[3];    // (2,B,K)
  const int*   role_idx     = (const int*)d_in[4];    // (2,2,B)
  const int*   neg_main     = (const int*)d_in[5];    // (2,B,50)
  const int*   neg_node     = (const int*)d_in[6];    // (2,2,B,10)
  const int*   neg_cross    = (const int*)d_in[7];    // (2,2,B,50)
  const int*   neg_role     = (const int*)d_in[8];    // (2,2,B,10)
  float*       partials     = (float*)d_ws;           // NTERMS*NBLK floats

  const float S = -1.0f / 10.0f;  // -(1/len(cost)), len == 10
  auto posc = [&](float w, int M) { return S * w / (float)(BATCH * M); };
  auto negc = [&](float w)        { return S * w / (float)BATCH; };

  JobArgs ja;
  int t = 0;
  for (int i = 0; i < NNET; ++i) {
    const float* NTi  = node_tables  + (size_t)i * VSZ * DIM;
    const float* NGTi = neigh_tables + (size_t)i * VSZ * DIM;
    const int*   ni   = nodes_idx + (size_t)i * BATCH;
    auto add = [&](const float* ctx, const int* idx, int M, int neg, float cf) {
      ja.jobs[t++] = Job{ctx, idx, NTi, ni, M, neg, cf, 0};
    };

    // main SGNS on view i
    add(NGTi, neigh_idx + (size_t)i * BATCH * KNB, KNB, 0, posc(1.0f, KNB));
    add(NGTi, neg_main + (size_t)i * BATCH * (KNB * NS), KNB * NS, 1, negc(1.0f));

    const int j = 1 - i;  // NUM_NET==2: the single j != i
    const float* NTj  = node_tables  + (size_t)j * VSZ * DIM;
    const float* NGTj = neigh_tables + (size_t)j * VSZ * DIM;

    // cross-view node alignment (HYP1)
    add(NTj, ni, 1, 0, posc(0.1f, 1));
    add(NTj, neg_node + ((size_t)i * NNET + j) * BATCH * NS, NS, 1, negc(0.1f));

    // cross-view neighbor loss (HYP2)
    add(NGTj, neigh_idx + (size_t)i * BATCH * KNB, KNB, 0, posc(0.1f, KNB));
    add(NGTj, neg_cross + ((size_t)i * NNET + j) * BATCH * (KNB * NS),
        KNB * NS, 1, negc(0.1f));

    // role-based loss over ALL j (HYP3)
    for (int jj = 0; jj < NNET; ++jj) {
      const float* NTjj = node_tables + (size_t)jj * VSZ * DIM;
      add(NTjj, role_idx + ((size_t)i * NNET + jj) * BATCH, 1, 0, posc(0.1f, 1));
      add(NTjj, neg_role + ((size_t)i * NNET + jj) * BATCH * NS, NS, 1,
          negc(0.1f));
    }
  }
  // t == NTERMS == 20

  terms_kernel<<<dim3(NBLK, NTERMS), TPB, 0, stream>>>(ja, partials);
  finalize_kernel<<<1, 32, 0, stream>>>(partials, (float*)d_out);
}
